// CSMultiHeadAttention_72696616452693
// MI455X (gfx1250) — matrix operand order repaired and checked by reading, not compiled, not run on hardware
//
#include <hip/hip_runtime.h>

// Problem dims
#define BB 8      // batch
#define SS 3072   // total seq
#define EE 512    // embedding
#define HH 8      // heads
#define NN 1024   // chunk length
#define DD 64     // head dim

typedef __bf16  bf16x16 __attribute__((ext_vector_type(16)));
typedef float   floatx8 __attribute__((ext_vector_type(8)));
typedef unsigned int uintx4 __attribute__((ext_vector_type(4)));

#if defined(__AMDGCN__) && __has_builtin(__builtin_amdgcn_tensor_load_to_lds)
#define USE_TDM 1
#else
#define USE_TDM 0
#endif

union Frag {
  bf16x16 v;
  uintx4  q[2];
};

__device__ __forceinline__ unsigned short f32_to_bf16(float f) {
  unsigned int u = __float_as_uint(f);
  u += 0x7FFFu + ((u >> 16) & 1u);   // round to nearest even
  return (unsigned short)(u >> 16);
}

__device__ __forceinline__ floatx8 wmma_bf16(const Frag& a, const Frag& b, floatx8 c) {
  return __builtin_amdgcn_wmma_f32_16x16x32_bf16(false, a.v, false, b.v,
                                                 (short)0, c, false, false);
}

// A-fragment (16x32 bf16) from row-major [M][K]; p -> tile origin (row0,k0).
// lanes<16: row=lr, K {0..7}&{16..23}; lanes>=16: row=lr, K {8..15}&{24..31}
__device__ __forceinline__ Frag load_afrag(const unsigned short* p, int ld, int lr, int half) {
  Frag f;
  const unsigned short* r = p + (size_t)lr * ld + half * 8;
  f.q[0] = *(const uintx4*)(r);
  f.q[1] = *(const uintx4*)(r + 16);
  return f;
}

// B-fragment (32x16 bf16) where Bt is row-major [N][K]; p -> tile origin (n0,k0).
// lanes<16: col N=lr, K {0..7}&{16..23}; lanes>=16: col N=lr, K {8..15}&{24..31}
__device__ __forceinline__ Frag load_bfrag(const unsigned short* p, int ld, int lr, int half) {
  Frag f;
  const unsigned short* r = p + (size_t)lr * ld + half * 8;
  f.q[0] = *(const uintx4*)(r);
  f.q[1] = *(const uintx4*)(r + 16);
  return f;
}

#if USE_TDM
typedef unsigned int u32x4 __attribute__((ext_vector_type(4)));
typedef int          i32x8 __attribute__((ext_vector_type(8)));
typedef int          i32x4 __attribute__((ext_vector_type(4)));

// 2-D bf16 tile DMA: global (row-major, row pitch in elements) -> LDS (packed tile).
// D# per CDNA5 ISA ch.8: group0 {count=1, lds_addr, global_addr, type=2},
// group1 {data_size=2B, tensor_dim0/1 = tile dims (no OOB), tile_dim0/1, dim0 stride}.
// This toolchain exposes the 6-arg builtin: (g0, g1, g2, g3, extra, cpol).
__device__ __forceinline__ void tdm_load_2d(unsigned int lds_off, const unsigned short* gptr,
                                            unsigned int tile_w, unsigned int tile_h,
                                            unsigned int row_stride_elems) {
  unsigned long long ga = (unsigned long long)(size_t)gptr;
  u32x4 g0;
  g0[0] = 1u;                                                  // count=1, user desc
  g0[1] = lds_off;                                             // LDS byte address
  g0[2] = (unsigned int)ga;                                    // global_addr[31:0]
  g0[3] = (unsigned int)((ga >> 32) & 0x01ffffffu) | 0x80000000u; // addr[56:32] | type=2
  i32x8 g1;
  g1[0] = 0x00010000;                                          // data_size = 2 bytes
  g1[1] = (int)(tile_w << 16);                                 // tensor_dim0[15:0]
  g1[2] = (int)(tile_h << 16);                                 // dim0 hi=0 | tensor_dim1[15:0]
  g1[3] = (int)(tile_w << 16);                                 // dim1 hi=0 | tile_dim0
  g1[4] = (int)tile_h;                                         // tile_dim1 (tile_dim2=0)
  g1[5] = (int)row_stride_elems;                               // tensor_dim0_stride lo
  g1[6] = 0;                                                   // stride hi | dim1_stride lo
  g1[7] = 0;
  i32x4 z4 = {0, 0, 0, 0};
  i32x8 z8 = {0, 0, 0, 0, 0, 0, 0, 0};
  __builtin_amdgcn_tensor_load_to_lds(g0, g1, z4, z4, z8, 0);
}

__device__ __forceinline__ void wait_tensorcnt0() {
#if __has_builtin(__builtin_amdgcn_s_wait_tensorcnt)
  __builtin_amdgcn_s_wait_tensorcnt((short)0);
#else
  asm volatile("s_wait_tensorcnt 0x0" ::: "memory");
#endif
}
#else
// Fallback: cooperative global -> LDS tile copy (uint4 granularity)
__device__ __forceinline__ void coop_copy_tile(unsigned short* dst, const unsigned short* src,
                                               int rows, int cols, int ld, int tid) {
  const int vpr = cols >> 3;          // uint4 vectors per row
  const int vecs = rows * vpr;
  for (int i = tid; i < vecs; i += 128) {
    const int r = i / vpr, cv = i - r * vpr;
    *(uintx4*)(dst + r * cols + cv * 8) = *(const uintx4*)(src + (size_t)r * ld + cv * 8);
  }
}
#endif

// ---------------- f32 -> bf16 conversion (vectorized x4) ----------------
struct __align__(8) us4 { unsigned short x, y, z, w; };

__global__ void cvt4_kernel(const float4* __restrict__ in, us4* __restrict__ out, int n4) {
  int i = blockIdx.x * blockDim.x + threadIdx.x;
  if (i >= n4) return;
  float4 f = in[i];
  us4 o;
  o.x = f32_to_bf16(f.x); o.y = f32_to_bf16(f.y);
  o.z = f32_to_bf16(f.z); o.w = f32_to_bf16(f.w);
  out[i] = o;
}

// ---------------- Generic Linear:  Y = X @ W^T + b  ----------------
// mode 0: Q -> bf16 [c][b][n][E]
// mode 1: K -> bf16 [c][b][n][E]
// mode 2: V -> bf16 transposed [c][b][h][d][n]
// mode 3: proj -> f32 d_out [b][S][E]   (X is attention output [c][b][n][E])
__global__ __launch_bounds__(128) void gemm_kernel(
    const unsigned short* __restrict__ X,
    const unsigned short* __restrict__ Wt,    // bf16 [3][E][E], row n (out), col k (in)
    const float* __restrict__ bias,           // [3][E]
    void* __restrict__ Out,
    int mode)
{
  const int c     = blockIdx.z;
  const int mtile = blockIdx.y;               // 0..127  (M = B*NN = 8192)
  const int ntile = blockIdx.x;               // 0..7
  const int w     = threadIdx.x >> 5;
  const int lane  = threadIdx.x & 31;
  const int lr    = lane & 15;
  const int half  = lane >> 4;

  const int mWG  = mtile * 64;
  const int n0   = ntile * 64;
  const int bWG  = mWG >> 10;                 // whole 64-row tile shares one batch block
  const int tokW = mWG & (NN - 1);

  const unsigned short* xrowWG =
      (mode == 3) ? X + (((size_t)(c * BB + bWG) << 10) + tokW) * EE
                  : X + ((size_t)bWG * SS + (size_t)c * NN + tokW) * EE;
  const unsigned short* wb = Wt + (size_t)c * EE * EE + (size_t)n0 * EE;
  const float* bb = bias + c * EE;

  __shared__ __align__(16) unsigned short xt[2][64][32];   // X tile (double buffered)
  __shared__ __align__(16) unsigned short wt[2][64][32];   // W tile (double buffered)

  auto issue_tiles = [&](int buf, int k0) {
#if USE_TDM
    if (w == 0) {
      tdm_load_2d((unsigned int)(size_t)&xt[buf][0][0], xrowWG + k0, 32, 64, EE);
      tdm_load_2d((unsigned int)(size_t)&wt[buf][0][0], wb + k0,     32, 64, EE);
    }
#else
    coop_copy_tile(&xt[buf][0][0], xrowWG + k0, 64, 32, EE, threadIdx.x);
    coop_copy_tile(&wt[buf][0][0], wb + k0,     64, 32, EE, threadIdx.x);
#endif
  };

  issue_tiles(0, 0);
#if USE_TDM
  if (w == 0) wait_tensorcnt0();
#endif
  __syncthreads();

  floatx8 vz = {0.f, 0.f, 0.f, 0.f, 0.f, 0.f, 0.f, 0.f};
  floatx8 acc[4] = {vz, vz, vz, vz};

  for (int k0 = 0; k0 < EE; k0 += 32) {
    const int cur = (k0 >> 5) & 1;
    if (k0 + 32 < EE) issue_tiles(cur ^ 1, k0 + 32);     // DMA next tile during compute

    // batch all fragment loads (one LDS clause), then stream 4 WMMAs
    Frag a   = load_afrag(&xt[cur][w * 16][0], 32, lr, half);
    Frag bf0 = load_bfrag(&wt[cur][ 0][0], 32, lr, half);
    Frag bf1 = load_bfrag(&wt[cur][16][0], 32, lr, half);
    Frag bf2 = load_bfrag(&wt[cur][32][0], 32, lr, half);
    Frag bf3 = load_bfrag(&wt[cur][48][0], 32, lr, half);
    acc[0] = wmma_bf16(a, bf0, acc[0]);
    acc[1] = wmma_bf16(a, bf1, acc[1]);
    acc[2] = wmma_bf16(a, bf2, acc[2]);
    acc[3] = wmma_bf16(a, bf3, acc[3]);
#if USE_TDM
    if (w == 0) wait_tensorcnt0();
#endif
    __syncthreads();
  }

  const int tok = tokW + w * 16;
#pragma unroll
  for (int t = 0; t < 4; t++) {
    const int ncol = n0 + t * 16 + lr;
    const float bv = bb[ncol];
#pragma unroll
    for (int r = 0; r < 8; r++) {
      const int tokl = tok + r + 8 * half;
      const float val = acc[t][r] + bv;
      if (mode == 3) {
        ((float*)Out)[((size_t)bWG * SS + (size_t)c * NN + tokl) * EE + ncol] = val;
      } else if (mode == 2) {
        const int h = ncol >> 6, dd = ncol & 63;
        ((unsigned short*)Out)[((((size_t)(c * BB + bWG)) * HH + h) * DD + dd) * NN + tokl] =
            f32_to_bf16(val);
      } else {
        ((unsigned short*)Out)[(((size_t)(c * BB + bWG)) * NN + tokl) * EE + ncol] =
            f32_to_bf16(val);
      }
    }
  }
}

// ---------------- Flash-style rotated cross-chunk attention ----------------
// out chunk cz: Q from chunk (cz+1)%3, K/V from chunk (cz+2)%3.
// 4 waves/WG share (c,b,h); K and V^T kv-tiles staged once per WG via TDM into
// double-buffered LDS. Per wave: one 16-query tile; S^T = K @ Q^T so softmax
// reductions are per-lane + one xor-16 shuffle; O^T = V^T @ P^T in f32 C-tiles.
__global__ __launch_bounds__(128) void attn_kernel(
    const unsigned short* __restrict__ Q,    // bf16 [c][b][n][E]
    const unsigned short* __restrict__ K,    // bf16 [c][b][n][E]
    const unsigned short* __restrict__ VT,   // bf16 [c][b][h][d][n]
    unsigned short* __restrict__ AO)         // bf16 [c][b][n][E]
{
  const int cz = blockIdx.z;
  const int bh = blockIdx.y;
  const int b = bh >> 3, h = bh & 7;
  const int w = threadIdx.x >> 5;
  const int q0 = (blockIdx.x * 4 + w) * 16;
  const int lane = threadIdx.x & 31;
  const int lr = lane & 15;
  const int half = lane >> 4;

  const int qc = (cz + 1) % 3;
  const int kc = (cz + 2) % 3;

  const unsigned short* qb  = Q  + (((size_t)(qc * BB + b)) * NN) * EE + h * DD;
  const unsigned short* kb  = K  + (((size_t)(kc * BB + b)) * NN) * EE + h * DD;
  const unsigned short* vtb = VT + ((((size_t)(kc * BB + b)) * HH + h) * DD) * NN;

  __shared__ __align__(16) unsigned short kt[2][32][64];   // K kv-tile   [j][d]
  __shared__ __align__(16) unsigned short vt[2][64][32];   // V^T kv-tile [d][j]
  __shared__ __align__(16) unsigned short pbuf[4][16][32]; // per-wave P^T staging

  auto issue_tiles = [&](int buf, int j0) {
#if USE_TDM
    if (w == 0) {
      tdm_load_2d((unsigned int)(size_t)&kt[buf][0][0], kb + (size_t)j0 * EE, 64, 32, EE);
      tdm_load_2d((unsigned int)(size_t)&vt[buf][0][0], vtb + j0,             32, 64, NN);
    }
#else
    coop_copy_tile(&kt[buf][0][0], kb + (size_t)j0 * EE, 32, 64, EE, threadIdx.x);
    coop_copy_tile(&vt[buf][0][0], vtb + j0,             64, 32, NN, threadIdx.x);
#endif
  };

  // Q B-fragments (N = query token, K = d), resident for the whole kv loop
  Frag bq0 = load_bfrag(qb + (size_t)q0 * EE + 0,  EE, lr, half);
  Frag bq1 = load_bfrag(qb + (size_t)q0 * EE + 32, EE, lr, half);

  issue_tiles(0, 0);
#if USE_TDM
  if (w == 0) wait_tensorcnt0();
#endif
  __syncthreads();

  floatx8 vz = {0.f, 0.f, 0.f, 0.f, 0.f, 0.f, 0.f, 0.f};
  floatx8 o[4] = {vz, vz, vz, vz};
  float mrun = -1e30f, lrun = 0.f;
  const float scale = 0.04419417382415922f;   // 1/sqrt(512)

  for (int j0 = 0; j0 < NN; j0 += 32) {
    const int cur = (j0 >> 5) & 1;
    if (j0 + 32 < NN) issue_tiles(cur ^ 1, j0 + 32);     // DMA next kv tile

    // S^T tiles from LDS: batch the 4 K-fragment loads, interleave the two
    // independent accumulator chains.
    Frag ka0 = load_afrag(&kt[cur][ 0][ 0], 64, lr, half);
    Frag ka1 = load_afrag(&kt[cur][ 0][32], 64, lr, half);
    Frag ka2 = load_afrag(&kt[cur][16][ 0], 64, lr, half);
    Frag ka3 = load_afrag(&kt[cur][16][32], 64, lr, half);
    floatx8 s0 = vz, s1 = vz;
    s0 = wmma_bf16(ka0, bq0, s0);
    s1 = wmma_bf16(ka2, bq0, s1);
    s0 = wmma_bf16(ka1, bq1, s0);
    s1 = wmma_bf16(ka3, bq1, s1);

    // online softmax over j (raw-domain max; scale folded into exp via fma)
    float tmax = -1e30f;
#pragma unroll
    for (int r = 0; r < 8; r++) tmax = fmaxf(tmax, fmaxf(s0[r], s1[r]));
    tmax = fmaxf(tmax, __shfl_xor(tmax, 16));
    const float mnew  = fmaxf(mrun, tmax * scale);
    const float alpha = __expf(mrun - mnew);
    float tsum = 0.f;
#pragma unroll
    for (int r = 0; r < 8; r++) {
      s0[r] = __expf(fmaf(s0[r], scale, -mnew));
      s1[r] = __expf(fmaf(s1[r], scale, -mnew));
      tsum += s0[r] + s1[r];
    }
    tsum += __shfl_xor(tsum, 16);
    lrun = lrun * alpha + tsum;
    mrun = mnew;
#pragma unroll
    for (int t = 0; t < 4; t++)
#pragma unroll
      for (int r = 0; r < 8; r++) o[t][r] *= alpha;

    // C-layout -> B-fragment layout via LDS: store P as [q][j]
#pragma unroll
    for (int r = 0; r < 8; r += 2) {
      unsigned int p0 = (unsigned int)f32_to_bf16(s0[r]) |
                        ((unsigned int)f32_to_bf16(s0[r + 1]) << 16);
      unsigned int p1 = (unsigned int)f32_to_bf16(s1[r]) |
                        ((unsigned int)f32_to_bf16(s1[r + 1]) << 16);
      *(unsigned int*)&pbuf[w][lr][8 * half + r]      = p0;
      *(unsigned int*)&pbuf[w][lr][16 + 8 * half + r] = p1;
    }
    asm volatile("s_wait_dscnt 0x0" ::: "memory");   // same-wave LDS RAW fence

    // batch P + V^T fragment loads, then 4 independent WMMA accumulations
    Frag pb  = load_bfrag(&pbuf[w][0][0], 32, lr, half);
    Frag va0 = load_afrag(&vt[cur][ 0][0], 32, lr, half);
    Frag va1 = load_afrag(&vt[cur][16][0], 32, lr, half);
    Frag va2 = load_afrag(&vt[cur][32][0], 32, lr, half);
    Frag va3 = load_afrag(&vt[cur][48][0], 32, lr, half);
    o[0] = wmma_bf16(va0, pb, o[0]);
    o[1] = wmma_bf16(va1, pb, o[1]);
    o[2] = wmma_bf16(va2, pb, o[2]);
    o[3] = wmma_bf16(va3, pb, o[3]);

#if USE_TDM
    if (w == 0) wait_tensorcnt0();
#endif
    __syncthreads();
  }

  const float inv = 1.f / lrun;
  unsigned short* aob =
      AO + (((size_t)(cz * BB + b)) * NN + q0 + lr) * EE + h * DD;
#pragma unroll
  for (int t = 0; t < 4; t++) {
#pragma unroll
    for (int r = 0; r < 8; r += 2) {
      const int dd = t * 16 + 8 * half + r;
      unsigned int pv = (unsigned int)f32_to_bf16(o[t][r] * inv) |
                        ((unsigned int)f32_to_bf16(o[t][r + 1] * inv) << 16);
      *(unsigned int*)(aob + dd) = pv;
    }
  }
}

// ---------------- Host launch ----------------
extern "C" void kernel_launch(void* const* d_in, const int* in_sizes, int n_in,
                              void* d_out, int out_size, void* d_ws, size_t ws_size,
                              hipStream_t stream) {
  const float* x  = (const float*)d_in[0];
  const float* Wq = (const float*)d_in[1];
  const float* bq = (const float*)d_in[2];
  const float* Wk = (const float*)d_in[3];
  const float* bk = (const float*)d_in[4];
  const float* Wv = (const float*)d_in[5];
  const float* bv = (const float*)d_in[6];
  const float* Wp = (const float*)d_in[7];
  const float* bp = (const float*)d_in[8];

  char* ws = (char*)d_ws;
  size_t off = 0;
  auto alloc = [&](size_t bytes) -> void* {
    void* p = ws + off;
    off = (off + bytes + 255) & ~(size_t)255;
    return p;
  };

  const size_t NXE = (size_t)BB * SS * EE;   // 12,582,912 elements
  const size_t NW  = (size_t)3 * EE * EE;    //    786,432 elements

  unsigned short* xb  = (unsigned short*)alloc(NXE * 2);
  unsigned short* wqb = (unsigned short*)alloc(NW * 2);
  unsigned short* wkb = (unsigned short*)alloc(NW * 2);
  unsigned short* wvb = (unsigned short*)alloc(NW * 2);
  unsigned short* wpb = (unsigned short*)alloc(NW * 2);
  unsigned short* qb  = (unsigned short*)alloc(NXE * 2);
  unsigned short* kbf = (unsigned short*)alloc(NXE * 2);
  unsigned short* vtb = (unsigned short*)alloc(NXE * 2);
  unsigned short* aob = (unsigned short*)alloc(NXE * 2);
  (void)ws_size; (void)in_sizes; (void)n_in; (void)out_size;

  // 1) convert activations + weights to bf16
  {
    int n4 = (int)(NXE / 4);
    cvt4_kernel<<<(n4 + 255) / 256, 256, 0, stream>>>((const float4*)x, (us4*)xb, n4);
    int w4 = (int)(NW / 4);
    cvt4_kernel<<<(w4 + 255) / 256, 256, 0, stream>>>((const float4*)Wq, (us4*)wqb, w4);
    cvt4_kernel<<<(w4 + 255) / 256, 256, 0, stream>>>((const float4*)Wk, (us4*)wkb, w4);
    cvt4_kernel<<<(w4 + 255) / 256, 256, 0, stream>>>((const float4*)Wv, (us4*)wvb, w4);
    cvt4_kernel<<<(w4 + 255) / 256, 256, 0, stream>>>((const float4*)Wp, (us4*)wpb, w4);
  }

  // 2) QKV projections (WMMA bf16, TDM-staged LDS tiles)
  dim3 gg(EE / 64, (BB * NN) / 64, 3), tb(128);
  gemm_kernel<<<gg, tb, 0, stream>>>(xb, wqb, bq, (void*)qb,  0);
  gemm_kernel<<<gg, tb, 0, stream>>>(xb, wkb, bk, (void*)kbf, 1);
  gemm_kernel<<<gg, tb, 0, stream>>>(xb, wvb, bv, (void*)vtb, 2);

  // 3) rotated cross-chunk flash attention
  dim3 ga(NN / 64, BB * HH, 3);
  attn_kernel<<<ga, tb, 0, stream>>>(qb, kbf, vtb, aob);

  // 4) output projection -> f32 d_out
  gemm_kernel<<<gg, tb, 0, stream>>>(aob, wpb, bp, d_out, 3);
}
